// BiLSTM_CRF_8315056685195
// MI455X (gfx1250) — compile-verified, matrix operand order checked by reading, not run on hardware
//
#include <hip/hip_runtime.h>
#include <hip/hip_bf16.h>

typedef __attribute__((ext_vector_type(16))) _Float16 v16h;
typedef __attribute__((ext_vector_type(8)))  float    v8f;

#define LQ   4096      // sequence length
#define EQ   100       // embed dim
#define HQ   128       // hidden
#define TQ   6         // tags
#define G3H  384       // 3*H
#define NCOL 768       // 3H * 2 directions
#define KPAD 128       // E padded to multiple of 32
#define MTILES 256     // LQ/16
#define NTILES 48      // NCOL/16
#define KTILES 4       // KPAD/32

// ---------------------------------------------------------------------------
// Prep A: gather embeddings for the sentence, convert to f16, and pre-swizzle
// into the WMMA 16-bit A-matrix per-lane layout:
//   lane<16: M=lane,    halves j<8 -> K=j,    j>=8 -> K=j+8
//   lane>=16: M=lane-16, same but K += 8
// Stored as [mtile][ktile][lane][16] halves so each lane loads one v16h.
// ---------------------------------------------------------------------------
__global__ void prep_a_kernel(const int* __restrict__ sent,
                              const float* __restrict__ emb,
                              _Float16* __restrict__ Aswz) {
    int tid   = blockIdx.x * blockDim.x + threadIdx.x;   // 524288 total
    int j     = tid & 15;
    int lane  = (tid >> 4) & 31;
    int kt    = (tid >> 9) & 3;
    int mtile = tid >> 11;
    int row   = mtile * 16 + (lane & 15);
    int khi   = lane >> 4;
    int K     = (j < 8 ? j : j + 8) + 8 * khi;
    int kk    = kt * 32 + K;
    float v = 0.0f;
    if (kk < EQ) {
        long long r = sent[row];
        v = emb[r * EQ + kk];
    }
    Aswz[tid] = (_Float16)v;
}

// ---------------------------------------------------------------------------
// Prep B: B[k][n] = Wih[n][k] (so gi = x @ Wih.T), both directions fused into
// N=768 columns, K padded 100->128, pre-swizzled analogously for v16h loads.
// ---------------------------------------------------------------------------
__global__ void prep_b_kernel(const float* __restrict__ w_ih_f,
                              const float* __restrict__ w_ih_b,
                              _Float16* __restrict__ Bswz) {
    int tid   = blockIdx.x * blockDim.x + threadIdx.x;   // 98304 total
    int j     = tid & 15;
    int lane  = (tid >> 4) & 31;
    int kt    = (tid >> 9) & 3;
    int ntile = tid >> 11;                               // 0..47
    int n     = ntile * 16 + (lane & 15);
    int khi   = lane >> 4;
    int K     = (j < 8 ? j : j + 8) + 8 * khi;
    int kk    = kt * 32 + K;
    float v = 0.0f;
    if (kk < EQ) {
        v = (n < G3H) ? w_ih_f[n * EQ + kk] : w_ih_b[(n - G3H) * EQ + kk];
    }
    Bswz[tid] = (_Float16)v;
}

// ---------------------------------------------------------------------------
// Prep WhhT: transpose recurrent weights so per-step global/LDS reads are
// coalesced and register preload is simple: WhhT[dir][k][j] = Whh[dir][j][k].
// ---------------------------------------------------------------------------
__global__ void prep_whh_kernel(const float* __restrict__ w_hh_f,
                                const float* __restrict__ w_hh_b,
                                float* __restrict__ WhhT) {
    int tid = blockIdx.x * blockDim.x + threadIdx.x;     // 98304 total
    int j   = tid % G3H;
    int rem = tid / G3H;
    int k   = rem % HQ;
    int dir = rem / HQ;
    const float* w = dir ? w_hh_b : w_hh_f;
    WhhT[(dir * HQ + k) * G3H + j] = w[j * HQ + k];
}

// ---------------------------------------------------------------------------
// WMMA GEMM: gi[L, 768] = x_pad[L,128] @ B[128,768] + bias
// One wave per 16x16 output tile; 8 waves per block; K loop of 4 WMMAs.
// ---------------------------------------------------------------------------
__global__ void gemm_gi_kernel(const _Float16* __restrict__ Aswz,
                               const _Float16* __restrict__ Bswz,
                               const float* __restrict__ b_ih_f,
                               const float* __restrict__ b_ih_b,
                               float* __restrict__ gi) {
    int wave  = threadIdx.x >> 5;
    int lane  = threadIdx.x & 31;
    int mtile = blockIdx.x / (NTILES / 8);
    int ntile = (blockIdx.x % (NTILES / 8)) * 8 + wave;

    v8f c = {};
    #pragma unroll
    for (int kt = 0; kt < KTILES; ++kt) {
        v16h a = *(const v16h*)(Aswz + (((size_t)mtile * KTILES + kt) * 32 + lane) * 16);
        v16h b = *(const v16h*)(Bswz + (((size_t)ntile * KTILES + kt) * 32 + lane) * 16);
        c = __builtin_amdgcn_wmma_f32_16x16x32_f16(
                /*neg_a=*/false, a, /*neg_b=*/false, b,
                /*c_mod=*/(short)0, c, /*reuse_a=*/false, /*reuse_b=*/false);
    }

    int n    = ntile * 16 + (lane & 15);
    float bi = (n < G3H) ? b_ih_f[n] : b_ih_b[n - G3H];
    int mbase = mtile * 16 + ((lane >> 4) << 3);
    #pragma unroll
    for (int r = 0; r < 8; ++r) {
        gi[(size_t)(mbase + r) * NCOL + n] = c[r] + bi;
    }
}

// ---------------------------------------------------------------------------
// Persistent GRU scan: blockIdx.x = direction (0 fwd, 1 bwd). 384 threads.
// Thread j owns output-gate row j; its 128 Whh weights live in registers.
// h lives in LDS (broadcast reads). Two barriers per step.
// ---------------------------------------------------------------------------
__global__ void __launch_bounds__(384, 1)
gru_scan_kernel(const float* __restrict__ gi,
                const float* __restrict__ WhhT,
                const float* __restrict__ h0,
                const float* __restrict__ b_hh_f,
                const float* __restrict__ b_hh_b,
                float* __restrict__ hs) {
    const int dir = blockIdx.x;
    const int j   = threadIdx.x;

    __shared__ float h_s[HQ];
    __shared__ float gh_s[G3H];
    __shared__ float gi_s[G3H];

    float w[HQ];
    const float* wt = WhhT + (size_t)dir * HQ * G3H;
    #pragma unroll
    for (int k = 0; k < HQ; ++k) w[k] = wt[(size_t)k * G3H + j];
    const float bj = (dir ? b_hh_b : b_hh_f)[j];

    if (j < HQ) h_s[j] = h0[dir * HQ + j];
    __syncthreads();

    for (int s = 0; s < LQ; ++s) {
        int t = dir ? (LQ - 1 - s) : s;
        const float* girow = gi + (size_t)t * NCOL + dir * G3H;
        float gij = girow[j];
        if (s + 1 < LQ) {
            int tn = dir ? (t - 1) : (t + 1);
            __builtin_prefetch(gi + (size_t)tn * NCOL + dir * G3H + j, 0, 1);
        }

        float acc = bj;
        #pragma unroll
        for (int k = 0; k < HQ; ++k) acc = fmaf(w[k], h_s[k], acc);
        gh_s[j] = acc;
        gi_s[j] = gij;
        __syncthreads();

        if (j < HQ) {
            float r  = 1.0f / (1.0f + __expf(-(gi_s[j]          + gh_s[j])));
            float z  = 1.0f / (1.0f + __expf(-(gi_s[HQ + j]     + gh_s[HQ + j])));
            float n  = tanhf(gi_s[2 * HQ + j] + r * gh_s[2 * HQ + j]);
            float hn = (1.0f - z) * n + z * h_s[j];
            h_s[j] = hn;
            hs[(size_t)t * (2 * HQ) + dir * HQ + j] = hn;
        }
        __syncthreads();
    }
}

// ---------------------------------------------------------------------------
// Output projection + softmax: logits[t] = softmax(concat(hf,hb) @ Wout.T + b)
// One thread per sequence position (6x256 dot products, Wout is L2-resident).
// ---------------------------------------------------------------------------
__global__ void out_proj_kernel(const float* __restrict__ hs,
                                const float* __restrict__ w_out,
                                const float* __restrict__ b_out,
                                float* __restrict__ logits) {
    int t = blockIdx.x * blockDim.x + threadIdx.x;
    if (t >= LQ) return;
    float acc[TQ];
    #pragma unroll
    for (int i = 0; i < TQ; ++i) acc[i] = b_out[i];
    const float* hr = hs + (size_t)t * (2 * HQ);
    for (int k = 0; k < 2 * HQ; ++k) {
        float x = hr[k];
        #pragma unroll
        for (int i = 0; i < TQ; ++i) acc[i] = fmaf(w_out[i * (2 * HQ) + k], x, acc[i]);
    }
    float m = acc[0];
    #pragma unroll
    for (int i = 1; i < TQ; ++i) m = fmaxf(m, acc[i]);
    float e[TQ], ssum = 0.0f;
    #pragma unroll
    for (int i = 0; i < TQ; ++i) { e[i] = __expf(acc[i] - m); ssum += e[i]; }
    float inv = 1.0f / ssum;
    #pragma unroll
    for (int i = 0; i < TQ; ++i) logits[(size_t)t * TQ + i] = e[i] * inv;
}

// ---------------------------------------------------------------------------
// Viterbi (faithful to reference): v[i,j] = prev[j] + trans[i,j], so the
// column-wise max/argmax of `trans` is step-invariant. Trellis recursion is
// elementwise per tag; backtrace uses constant per-column backpointers.
// path (L+1 entries, as floats) then score -> d_out.
// ---------------------------------------------------------------------------
__global__ void viterbi_kernel(const float* __restrict__ logits,
                               const float* __restrict__ trans,
                               float* __restrict__ dout) {
    __shared__ float trell[TQ];
    __shared__ int   bpcol[TQ];
    int j = threadIdx.x;
    if (j < TQ) {
        float best = trans[0 * TQ + j];
        int   bi   = 0;
        for (int i = 1; i < TQ; ++i) {
            float v = trans[i * TQ + j];
            if (v > best) { best = v; bi = i; }
        }
        bpcol[j] = bi;
        float tr = logits[j];                       // t = 0
        for (int t = 1; t < LQ; ++t) tr = logits[(size_t)t * TQ + j] + tr + best;
        trell[j] = tr;
    }
    __syncthreads();
    if (j == 0) {
        int start = 0;
        float sc = trell[0];
        for (int i = 1; i < TQ; ++i) if (trell[i] > sc) { sc = trell[i]; start = i; }
        dout[LQ] = (float)start;                    // path[L] = start
        int cur = start;
        for (int row = LQ - 1; row >= 0; --row) {
            int nxt = (row == 0) ? 0 : bpcol[cur];  // bp row 0 is zeros
            dout[row] = (float)nxt;
            cur = nxt;
        }
        dout[LQ + 1] = sc;                          // score
    }
}

// ---------------------------------------------------------------------------
extern "C" void kernel_launch(void* const* d_in, const int* in_sizes, int n_in,
                              void* d_out, int out_size, void* d_ws, size_t ws_size,
                              hipStream_t stream) {
    const int*   sent    = (const int*)  d_in[0];
    const float* h0      = (const float*)d_in[1];
    const float* emb     = (const float*)d_in[2];
    const float* w_ih_f  = (const float*)d_in[3];
    const float* w_hh_f  = (const float*)d_in[4];
    const float* b_ih_f  = (const float*)d_in[5];
    const float* b_hh_f  = (const float*)d_in[6];
    const float* w_ih_b  = (const float*)d_in[7];
    const float* w_hh_b  = (const float*)d_in[8];
    const float* b_ih_b  = (const float*)d_in[9];
    const float* b_hh_b  = (const float*)d_in[10];
    const float* w_out   = (const float*)d_in[11];
    const float* b_out   = (const float*)d_in[12];
    const float* trans   = (const float*)d_in[13];
    float* dout = (float*)d_out;

    // Carve workspace (256B aligned chunks).
    char* ws = (char*)d_ws;
    size_t off = 0;
    auto carve = [&](size_t bytes) -> void* {
        void* p = ws + off;
        off = (off + bytes + 255) & ~(size_t)255;
        return p;
    };
    _Float16* Aswz  = (_Float16*)carve((size_t)LQ * KPAD * sizeof(_Float16));      // 1 MB
    _Float16* Bswz  = (_Float16*)carve((size_t)NCOL * KPAD * sizeof(_Float16));    // 192 KB
    float*    WhhT  = (float*)   carve((size_t)2 * HQ * G3H * sizeof(float));      // 384 KB
    float*    gi    = (float*)   carve((size_t)LQ * NCOL * sizeof(float));         // 12 MB
    float*    hs    = (float*)   carve((size_t)LQ * 2 * HQ * sizeof(float));       // 4 MB
    float*    logits= (float*)   carve((size_t)LQ * TQ * sizeof(float));           // 96 KB

    // 1) Pre-swizzle A (embedding gather) and B (input weights), transpose Whh.
    prep_a_kernel<<<(LQ * KPAD) / 256, 256, 0, stream>>>(sent, emb, Aswz);
    prep_b_kernel<<<(NCOL * KPAD) / 256, 256, 0, stream>>>(w_ih_f, w_ih_b, Bswz);
    prep_whh_kernel<<<(2 * HQ * G3H) / 256, 256, 0, stream>>>(w_hh_f, w_hh_b, WhhT);

    // 2) WMMA GEMM: gi = x_pad @ B + bias (both directions fused, N = 768).
    gemm_gi_kernel<<<MTILES * (NTILES / 8), 256, 0, stream>>>(Aswz, Bswz, b_ih_f, b_ih_b, gi);

    // 3) Persistent bidirectional GRU scan (2 workgroups, one per direction).
    gru_scan_kernel<<<2, G3H, 0, stream>>>(gi, WhhT, h0, b_hh_f, b_hh_b, hs);

    // 4) Output projection + softmax.
    out_proj_kernel<<<LQ / 256, 256, 0, stream>>>(hs, w_out, b_out, logits);

    // 5) Viterbi decode + backtrace.
    viterbi_kernel<<<1, 32, 0, stream>>>(logits, trans, dout);
}
